// DGCNN_34041910788657
// MI455X (gfx1250) — matrix operand order repaired and checked by reading, not compiled, not run on hardware
//
#include <hip/hip_runtime.h>
#include <hip/hip_bf16.h>

// ---------------------------------------------------------------------------
// DGCNN forward (B=16 events, N=2048 pts, k=20) for gfx1250 (MI455X).
// All GEMM-shaped work (edge MLPs, lin1, and the 64-D kNN distance matrix)
// runs on v_wmma_f32_16x16x32_f16 (wave32).  Weights are pre-swizzled into
// WMMA B-fragment order at f32->f16 conversion so every B fragment is a
// contiguous 32-byte load (2x ds_load_b128 / global_load_b128).
//
// d_in layout assumption (setup_inputs() insertion order, leaves flattened):
//  0: pos [B*N,3] f32
//  1..4  : conv1 L1  W[6,64]   b g beta [64]
//  5..8  : conv1 L2  W[64,64]  b g beta [64]
//  9..12 : conv1 L3  W[64,64]  b g beta [64]
// 13..16 : conv2     W[128,128] b g beta [128]
// 17..20 : lin1      W[192,1024] b g beta [1024]
// 21..24 : mlp1      W[1024,512] b g beta [512]
// 25..28 : mlp2      W[512,256]  b g beta [256]
// 29: headW [256,5]  30: headb [5]  31: batch (unused; equal-sized events)
// ---------------------------------------------------------------------------

typedef __attribute__((ext_vector_type(16))) _Float16 v16h;
typedef __attribute__((ext_vector_type(8)))  float    v8f;

#define B_EV 16
#define NPT  2048
#define KNN  20

#define WMMA16(a, bm, cm) \
  __builtin_amdgcn_wmma_f32_16x16x32_f16(false, (a), false, (bm), (short)0, (cm), false, false)

// ---- CDNA5 wave32 fragment loaders ----------------------------------------

// A fragment: 16x32 f16 from row-major LDS. lanes 0-15: row=lane, K=0..15;
// lanes 16-31: row=lane-16, K=16..31.  Two b128 LDS loads.
__device__ __forceinline__ v16h ld_afrag(const _Float16* base, int ld) {
  int l = threadIdx.x & 31;
  const _Float16* p = base + (l & 15) * ld + ((l >> 4) << 4);
  v16h a;
#pragma unroll
  for (int t = 0; t < 8; ++t) a[t] = p[t];
#pragma unroll
  for (int t = 0; t < 8; ++t) a[8 + t] = p[8 + t];
  return a;
}

// B fragment from pre-swizzled block: 512 halves per 32x16 (KxN) block,
// lane l's 16 halves contiguous at base + l*16.  (k = kb + t, n = l&15,
// kb = (l>=16)*16.)
__device__ __forceinline__ v16h ld_bfrag_sw(const _Float16* base) {
  int l = threadIdx.x & 31;
  const _Float16* p = base + l * 16;
  v16h b;
#pragma unroll
  for (int t = 0; t < 16; ++t) b[t] = p[t];
  return b;
}

// B fragment = transpose of a row-major [16 rows][ld] tile: column n of B is
// row n of the tile (B = tile^T, 32x16 from a 16x32 tile).  lane l: n = l&15,
// K = (l>=16)*16 + t, element = tile[n*ld + kb + t] -- 16 contiguous halves.
__device__ __forceinline__ v16h ld_bfrag_rm(const _Float16* base, int ld) {
  int l = threadIdx.x & 31;
  const _Float16* p = base + (l & 15) * ld + ((l >> 4) << 4);
  v16h b;
#pragma unroll
  for (int t = 0; t < 16; ++t) b[t] = p[t];
  return b;
}

// Layer: relu(x + b) * g + beta
__device__ __forceinline__ float actf(float x, float bb, float gg, float be) {
  float r = x + bb;
  r = r > 0.f ? r : 0.f;
  return r * gg + be;
}

// Apply activation to a C fragment and store f16 into LDS [rows][ldh].
// C/D layout: lane&15 = N col; VGPR r -> M = mt*16 + r (+8 for lanes>=16).
__device__ __forceinline__ void act_store16(_Float16* H, int ldh, int mt, int nt, v8f c,
                                            const float* bias, const float* g,
                                            const float* beta) {
  int l   = threadIdx.x & 31;
  int col = nt * 16 + (l & 15);
  float bb = bias[col], gg = g[col], be = beta[col];
  int mb = mt * 16 + ((l >> 4) << 3);
#pragma unroll
  for (int r = 0; r < 8; ++r)
    H[(mb + r) * ldh + col] = (_Float16)actf(c[r], bb, gg, be);
}

// ---- order-preserving float<->uint for deterministic atomic max pooling ---
__device__ __forceinline__ unsigned f2ord(float f) {
  unsigned u = __float_as_uint(f);
  return (u & 0x80000000u) ? ~u : (u | 0x80000000u);
}
__device__ __forceinline__ float ord2f(unsigned u) {
  unsigned v = (u & 0x80000000u) ? (u & 0x7FFFFFFFu) : ~u;
  return __uint_as_float(v);
}

// ---- top-k insertion (register-resident sorted list, matches jax top_k
//      tie behavior: strict < keeps lower candidate index first).
//      Inner loop MUST be fully unrolled (register indexing); callers must
//      NOT unroll their candidate loops or this code is cloned per iteration.
template <int KK>
__device__ __forceinline__ void knn_insert(float d, int j, float* bd, int* bi) {
  if (d >= bd[KK - 1]) return;
  float cd = d; int ci = j;
#pragma unroll
  for (int s = 0; s < KK; ++s) {
    if (cd < bd[s]) {
      float td = bd[s]; int ti = bi[s];
      bd[s] = cd; bi[s] = ci; cd = td; ci = ti;
    }
  }
}

// ===========================================================================
// utility kernels
// ===========================================================================
__global__ void dg_zero_u32(unsigned* p, int n) {
  int t = blockIdx.x * blockDim.x + threadIdx.x;
  if (t < n) p[t] = 0u;  // ord-encoding of the most negative value
}

// Convert f32 weights [Ksrc][N] (row-major, in x out) to f16 in WMMA
// B-fragment block order, zero-padding K up to Kpad (multiple of 32).
// Block (kc, nt) occupies 512 halves at ((kc*(N/16) + nt) * 512); within a
// block, element (lane, t) = W[kc*32 + (lane>=16)*16 + t][nt*16 + (lane&15)].
__global__ void dg_cvt_f16_sw(const float* __restrict__ src, _Float16* __restrict__ dst,
                              int Ksrc, int Kpad, int N) {
  int e = blockIdx.x * blockDim.x + threadIdx.x;
  if (e >= Kpad * N) return;
  int block  = e >> 9;
  int within = e & 511;
  int lane = within >> 4;
  int t    = within & 15;
  int nblocks = N >> 4;
  int nt = block % nblocks;
  int kc = block / nblocks;
  int k = kc * 32 + ((lane >> 4) << 4) + t;
  int n = nt * 16 + (lane & 15);
  dst[e] = (k < Ksrc) ? (_Float16)src[k * N + n] : (_Float16)0.f;
}

// ===========================================================================
// kNN on pos (D=3).  grid = B*8 blocks x 256 threads, 1 query/thread.
// D=3 is too shallow for the matrix pipes -- stays VALU (0.4 GFLOP).
// ===========================================================================
__global__ void __launch_bounds__(256) dg_knn3(const float* __restrict__ pos,
                                               int* __restrict__ idx) {
  __shared__ float sp[NPT * 3];  // 24 KB
  int b  = blockIdx.x >> 3;
  int qt = blockIdx.x & 7;
  const float* base = pos + (size_t)b * NPT * 3;
  for (int t = threadIdx.x; t < NPT * 3; t += 256) sp[t] = base[t];
  __syncthreads();

  int i = qt * 256 + threadIdx.x;
  float qx = sp[i * 3 + 0], qy = sp[i * 3 + 1], qz = sp[i * 3 + 2];
  float bd[KNN]; int bi[KNN];
#pragma unroll
  for (int s = 0; s < KNN; ++s) { bd[s] = 3.4e38f; bi[s] = 0; }

#pragma unroll 1
  for (int j = 0; j < NPT; ++j) {
    float dx = qx - sp[j * 3 + 0];
    float dy = qy - sp[j * 3 + 1];
    float dz = qz - sp[j * 3 + 2];
    knn_insert<KNN>(dx * dx + dy * dy + dz * dz, j, bd, bi);
  }
  int* o = idx + ((size_t)b * NPT + i) * KNN;
#pragma unroll
  for (int s = 0; s < KNN; ++s) o[s] = bi[s];
}

// ===========================================================================
// kNN on x1 (D=64) via WMMA distance matrix: d = |q|^2 + |c|^2 - 2 q.c
// (same algebraic form as the reference einsum path).
// grid = B*16 blocks x 256 threads (8 waves); each wave owns 16 queries,
// candidates staged 128 at a time as f16.  Per candidate sub-tile: 2 WMMAs
// produce a 16x16 dot tile; lanes 0-15 then scan one query-row each into a
// register-resident top-20 list.
// ===========================================================================
__global__ void __launch_bounds__(256) dg_knn64w(const float* __restrict__ x1,
                                                 int* __restrict__ idx) {
  __shared__ _Float16 Qh[128 * 64];    // 16 KB
  __shared__ _Float16 Ch[128 * 64];    // 16 KB
  __shared__ float    sqQ[128];
  __shared__ float    sqC[128];
  __shared__ float    Dt[8][16 * 16];  //  8 KB (wave-private dot tiles)

  int b   = blockIdx.x >> 4;
  int qt  = blockIdx.x & 15;
  int tid = threadIdx.x, w = tid >> 5, l = tid & 31;
  int qbase = qt * 128;

  // stage queries (f32 -> f16) + squared norms of the f16-rounded values
  for (int t = tid; t < 128 * 64; t += 256)
    Qh[t] = (_Float16)x1[((size_t)b * NPT + qbase) * 64 + t];
  __syncthreads();
  if (tid < 128) {
    float s = 0.f;
#pragma unroll 8
    for (int k = 0; k < 64; ++k) { float v = (float)Qh[tid * 64 + k]; s = fmaf(v, v, s); }
    sqQ[tid] = s;
  }

  float bd[KNN]; int bi[KNN];
#pragma unroll
  for (int s = 0; s < KNN; ++s) { bd[s] = 3.4e38f; bi[s] = 0; }

  float* Dw = Dt[w];
  const _Float16* Qw = Qh + (w * 16) * 64;

#pragma unroll 1
  for (int ct = 0; ct < 16; ++ct) {
    __syncthreads();
    for (int t = tid; t < 128 * 64; t += 256)
      Ch[t] = (_Float16)x1[((size_t)b * NPT + ct * 128) * 64 + t];
    __syncthreads();
    if (tid < 128) {
      float s = 0.f;
#pragma unroll 8
      for (int k = 0; k < 64; ++k) { float v = (float)Ch[tid * 64 + k]; s = fmaf(v, v, s); }
      sqC[tid] = s;
    }
    __syncthreads();

#pragma unroll 1
    for (int st = 0; st < 8; ++st) {
      v8f acc = {};
#pragma unroll
      for (int kc = 0; kc < 2; ++kc) {
        v16h a  = ld_afrag(Qw + kc * 32, 64);
        v16h bf = ld_bfrag_rm(Ch + (st * 16) * 64 + kc * 32, 64);  // C^T fragment
        acc = WMMA16(a, bf, acc);
      }
      int mb = (l >> 4) << 3;
#pragma unroll
      for (int r = 0; r < 8; ++r) Dw[(mb + r) * 16 + (l & 15)] = acc[r];
      __builtin_amdgcn_wave_barrier();  // LDS ops are in-order per wave
      if (l < 16) {
        float qs = sqQ[w * 16 + l];
        int jb = ct * 128 + st * 16;
#pragma unroll 1
        for (int n = 0; n < 16; ++n) {
          float d = qs + sqC[st * 16 + n] - 2.f * Dw[l * 16 + n];
          knn_insert<KNN>(d, jb + n, bd, bi);
        }
      }
      __builtin_amdgcn_wave_barrier();
    }
  }

  if (l < 16) {
    int* o = idx + ((size_t)b * NPT + qbase + w * 16 + l) * KNN;
#pragma unroll
    for (int s = 0; s < KNN; ++s) o[s] = bi[s];
  }
}

// ===========================================================================
// EdgeConv1: per point, edges [20x6] (padded 32x32) -> 64 -> 64 -> 64,
// then column max over the 20 valid rows.  1 wave per point, 4 waves/block.
// 40 WMMAs per point.  Weights arrive pre-swizzled (fragment order).
// ===========================================================================
__global__ void __launch_bounds__(128) dg_edgeconv1(
    const float* __restrict__ pos, const int* __restrict__ idx,
    const _Float16* __restrict__ W1h, const float* b1, const float* g1, const float* t1,
    const _Float16* __restrict__ W2h, const float* b2, const float* g2, const float* t2,
    const _Float16* __restrict__ W3h, const float* b3, const float* g3, const float* t3,
    float* __restrict__ x1out) {
  __shared__ _Float16 W1s[32 * 64];   //  4 KB (swizzled blocks, K padded)
  __shared__ _Float16 W2s[64 * 64];   //  8 KB (swizzled blocks)
  __shared__ _Float16 W3s[64 * 64];   //  8 KB (swizzled blocks)
  __shared__ _Float16 E[4][32 * 32];  //  8 KB
  __shared__ _Float16 H1[4][32 * 64]; // 16 KB
  __shared__ _Float16 H2[4][32 * 64]; // 16 KB  -> 60 KB total

  int tid = threadIdx.x;
  for (int t = tid; t < 32 * 64; t += 128) W1s[t] = W1h[t];
  for (int t = tid; t < 64 * 64; t += 128) W2s[t] = W2h[t];
  for (int t = tid; t < 64 * 64; t += 128) W3s[t] = W3h[t];

  int w = tid >> 5, l = tid & 31;
  int p = blockIdx.x * 4 + w;
  int b = p >> 11, i = p & (NPT - 1);

  _Float16* Ew  = E[w];
  _Float16* H1w = H1[w];
  _Float16* H2w = H2[w];
  for (int t = l; t < 32 * 32; t += 32) Ew[t] = (_Float16)0.f;
  __syncthreads();

  // gather edge features: rows = edges, e = [x_i, x_j - x_i] (6 dims)
  if (l < KNN) {
    int j = idx[((size_t)b * NPT + i) * KNN + l];
    const float* xi = pos + ((size_t)b * NPT + i) * 3;
    const float* xj = pos + ((size_t)b * NPT + j) * 3;
#pragma unroll
    for (int c = 0; c < 3; ++c) {
      float a = xi[c];
      Ew[l * 32 + c]     = (_Float16)a;
      Ew[l * 32 + 3 + c] = (_Float16)(xj[c] - a);
    }
  }
  __syncthreads();

  // layer 1: [32x32] x [32x64]   (blocks: nt = 0..3, kc = 0)
  for (int mt = 0; mt < 2; ++mt) {
    v16h a = ld_afrag(Ew + mt * 16 * 32, 32);
#pragma unroll
    for (int nt = 0; nt < 4; ++nt) {
      v16h bf = ld_bfrag_sw(W1s + nt * 512);
      v8f c = {};
      c = WMMA16(a, bf, c);
      act_store16(H1w, 64, mt, nt, c, b1, g1, t1);
    }
  }
  __syncthreads();

  // layer 2: [32x64] x [64x64]   (block (kc, nt) at (kc*4 + nt) * 512)
  for (int nt = 0; nt < 4; ++nt) {
    v8f c0 = {}, c1 = {};
#pragma unroll
    for (int kc = 0; kc < 2; ++kc) {
      v16h bf = ld_bfrag_sw(W2s + (kc * 4 + nt) * 512);
      v16h a0 = ld_afrag(H1w + kc * 32, 64);
      v16h a1 = ld_afrag(H1w + 16 * 64 + kc * 32, 64);
      c0 = WMMA16(a0, bf, c0);
      c1 = WMMA16(a1, bf, c1);
    }
    act_store16(H2w, 64, 0, nt, c0, b2, g2, t2);
    act_store16(H2w, 64, 1, nt, c1, b2, g2, t2);
  }
  __syncthreads();

  // layer 3 + max over rows 0..19
  for (int nt = 0; nt < 4; ++nt) {
    v8f c0 = {}, c1 = {};
#pragma unroll
    for (int kc = 0; kc < 2; ++kc) {
      v16h bf = ld_bfrag_sw(W3s + (kc * 4 + nt) * 512);
      v16h a0 = ld_afrag(H2w + kc * 32, 64);
      v16h a1 = ld_afrag(H2w + 16 * 64 + kc * 32, 64);
      c0 = WMMA16(a0, bf, c0);
      c1 = WMMA16(a1, bf, c1);
    }
    int col = nt * 16 + (l & 15);
    float bb = b3[col], gg = g3[col], be = t3[col];
    float cm = -3.4e38f;
#pragma unroll
    for (int r = 0; r < 8; ++r) cm = fmaxf(cm, actf(c0[r], bb, gg, be));  // rows 0..15
    int mb = 16 + ((l >> 4) << 3);
#pragma unroll
    for (int r = 0; r < 8; ++r) {  // rows 16..31: only <20 valid
      float v = actf(c1[r], bb, gg, be);
      if (mb + r < KNN) cm = fmaxf(cm, v);
    }
    cm = fmaxf(cm, __shfl_xor(cm, 16));
    if (l < 16) x1out[((size_t)b * NPT + i) * 64 + col] = cm;
  }
}

// ===========================================================================
// EdgeConv2: per point, edges [20x128] (padded 32x128) -> 128, max over rows.
// 1 wave per point, 2 waves/block.  64 WMMAs per point.
// ===========================================================================
__global__ void __launch_bounds__(64) dg_edgeconv2(
    const float* __restrict__ x1, const int* __restrict__ idx,
    const _Float16* __restrict__ Wh, const float* bias, const float* g,
    const float* beta, float* __restrict__ x2out) {
  __shared__ _Float16 Ws[128 * 128];  // 32 KB (swizzled blocks)
  __shared__ _Float16 E[2][32 * 128]; // 16 KB -> 48 KB total

  int tid = threadIdx.x;
  for (int t = tid; t < 128 * 128; t += 64) Ws[t] = Wh[t];

  int w = tid >> 5, l = tid & 31;
  int p = blockIdx.x * 2 + w;
  int b = p >> 11, i = p & (NPT - 1);

  _Float16* Ew = E[w];
  for (int t = l; t < 32 * 128; t += 32) Ew[t] = (_Float16)0.f;
  __syncthreads();

  if (l < KNN) {
    int j = idx[((size_t)b * NPT + i) * KNN + l];
    const float* xi = x1 + ((size_t)b * NPT + i) * 64;
    const float* xj = x1 + ((size_t)b * NPT + j) * 64;
#pragma unroll
    for (int c = 0; c < 64; ++c) {
      float a = xi[c];
      Ew[l * 128 + c]      = (_Float16)a;
      Ew[l * 128 + 64 + c] = (_Float16)(xj[c] - a);
    }
  }
  __syncthreads();

  for (int nt = 0; nt < 8; ++nt) {
    v8f c0 = {}, c1 = {};
#pragma unroll
    for (int kc = 0; kc < 4; ++kc) {
      v16h bf = ld_bfrag_sw(Ws + (kc * 8 + nt) * 512);
      v16h a0 = ld_afrag(Ew + kc * 32, 128);
      v16h a1 = ld_afrag(Ew + 16 * 128 + kc * 32, 128);
      c0 = WMMA16(a0, bf, c0);
      c1 = WMMA16(a1, bf, c1);
    }
    int col = nt * 16 + (l & 15);
    float bb = bias[col], gg = g[col], be = beta[col];
    float cm = -3.4e38f;
#pragma unroll
    for (int r = 0; r < 8; ++r) cm = fmaxf(cm, actf(c0[r], bb, gg, be));
    int mb = 16 + ((l >> 4) << 3);
#pragma unroll
    for (int r = 0; r < 8; ++r) {
      float v = actf(c1[r], bb, gg, be);
      if (mb + r < KNN) cm = fmaxf(cm, v);
    }
    cm = fmaxf(cm, __shfl_xor(cm, 16));
    if (l < 16) x2out[((size_t)b * NPT + i) * 128 + col] = cm;
  }
}

// ===========================================================================
// lin1 (192 -> 1024) fused with global max pool.
// grid = (nTile=8, pointTile=128, batch=16), 256 threads (8 waves).
// Block computes 16 points x 128 cols; wave w owns global col tile nb*8 + w.
// B fragments stream straight from L2 (swizzled layout -> coalesced b128s).
// ===========================================================================
__global__ void __launch_bounds__(256) dg_lin1pool(
    const float* __restrict__ x1, const float* __restrict__ x2,
    const _Float16* __restrict__ Wh, const float* bias, const float* g,
    const float* beta, unsigned* __restrict__ pooled) {
  __shared__ _Float16 A[16 * 192];  // 6 KB

  int nb = blockIdx.x;
  int pt = blockIdx.y;
  int b  = blockIdx.z;
  int tid = threadIdx.x;
  int w = tid >> 5, l = tid & 31;

  // stage A tile: concat [x1 | x2] for 16 points, f32 -> f16
  for (int t = tid; t < 16 * 192; t += 256) {
    int r = t / 192, c = t % 192;
    int p = pt * 16 + r;
    float v = (c < 64) ? x1[((size_t)b * NPT + p) * 64 + c]
                       : x2[((size_t)b * NPT + p) * 128 + (c - 64)];
    A[t] = (_Float16)v;
  }
  __syncthreads();

  int ntg = nb * 8 + w;  // global N tile 0..63
  v8f acc = {};
#pragma unroll
  for (int kc = 0; kc < 6; ++kc) {
    v16h a  = ld_afrag(A + kc * 32, 192);
    v16h bf = ld_bfrag_sw(Wh + (size_t)(kc * 64 + ntg) * 512);
    acc = WMMA16(a, bf, acc);
  }

  int col = ntg * 16 + (l & 15);
  float bb = bias[col], gg = g[col], be = beta[col];
  float cm = -3.4e38f;
#pragma unroll
  for (int r = 0; r < 8; ++r) cm = fmaxf(cm, actf(acc[r], bb, gg, be));
  cm = fmaxf(cm, __shfl_xor(cm, 16));
  if (l < 16) atomicMax(&pooled[(size_t)b * 1024 + col], f2ord(cm));
}

// ===========================================================================
// head: pooled[16,1024] -> mlp1(512) -> mlp2(256) -> head(5). 1 block/batch.
// ===========================================================================
__global__ void __launch_bounds__(256) dg_head(
    const unsigned* __restrict__ pooled,
    const float* m1W, const float* m1b, const float* m1g, const float* m1t,
    const float* m2W, const float* m2b, const float* m2g, const float* m2t,
    const float* hW, const float* hb, float* __restrict__ out) {
  __shared__ float P[1024];
  __shared__ float H1[512];
  __shared__ float H2[256];
  int b = blockIdx.x, tid = threadIdx.x;

  for (int t = tid; t < 1024; t += 256) P[t] = ord2f(pooled[(size_t)b * 1024 + t]);
  __syncthreads();

  for (int o = tid; o < 512; o += 256) {
    float s = 0.f;
#pragma unroll 8
    for (int k = 0; k < 1024; ++k) s = fmaf(P[k], m1W[(size_t)k * 512 + o], s);
    H1[o] = actf(s, m1b[o], m1g[o], m1t[o]);
  }
  __syncthreads();

  if (tid < 256) {
    float s = 0.f;
#pragma unroll 8
    for (int k = 0; k < 512; ++k) s = fmaf(H1[k], m2W[(size_t)k * 256 + tid], s);
    H2[tid] = actf(s, m2b[tid], m2g[tid], m2t[tid]);
  }
  __syncthreads();

  if (tid < 5) {
    float s = hb[tid];
#pragma unroll 8
    for (int k = 0; k < 256; ++k) s = fmaf(H2[k], hW[k * 5 + tid], s);
    out[b * 5 + tid] = s;
  }
}

// ===========================================================================
// launcher
// ===========================================================================
extern "C" void kernel_launch(void* const* d_in, const int* in_sizes, int n_in,
                              void* d_out, int out_size, void* d_ws, size_t ws_size,
                              hipStream_t stream) {
  (void)in_sizes; (void)n_in; (void)out_size; (void)ws_size;

  const float* pos  = (const float*)d_in[0];
  const float* c1W1 = (const float*)d_in[1];
  const float* c1b1 = (const float*)d_in[2];
  const float* c1g1 = (const float*)d_in[3];
  const float* c1t1 = (const float*)d_in[4];
  const float* c1W2 = (const float*)d_in[5];
  const float* c1b2 = (const float*)d_in[6];
  const float* c1g2 = (const float*)d_in[7];
  const float* c1t2 = (const float*)d_in[8];
  const float* c1W3 = (const float*)d_in[9];
  const float* c1b3 = (const float*)d_in[10];
  const float* c1g3 = (const float*)d_in[11];
  const float* c1t3 = (const float*)d_in[12];
  const float* c2W  = (const float*)d_in[13];
  const float* c2b  = (const float*)d_in[14];
  const float* c2g  = (const float*)d_in[15];
  const float* c2t  = (const float*)d_in[16];
  const float* lW   = (const float*)d_in[17];
  const float* lb   = (const float*)d_in[18];
  const float* lg   = (const float*)d_in[19];
  const float* lt   = (const float*)d_in[20];
  const float* m1W  = (const float*)d_in[21];
  const float* m1b  = (const float*)d_in[22];
  const float* m1g  = (const float*)d_in[23];
  const float* m1t  = (const float*)d_in[24];
  const float* m2W  = (const float*)d_in[25];
  const float* m2b  = (const float*)d_in[26];
  const float* m2g  = (const float*)d_in[27];
  const float* m2t  = (const float*)d_in[28];
  const float* hW   = (const float*)d_in[29];
  const float* hb   = (const float*)d_in[30];
  float* out = (float*)d_out;

  // workspace carve-out (256-byte aligned slices)
  char* ws = (char*)d_ws;
  size_t off = 0;
  auto take = [&](size_t bytes) -> char* {
    char* p = ws + off;
    off += (bytes + 255) & ~(size_t)255;
    return p;
  };
  int*       idx1   = (int*)take((size_t)B_EV * NPT * KNN * 4);
  int*       idx2   = (int*)take((size_t)B_EV * NPT * KNN * 4);
  float*     x1     = (float*)take((size_t)B_EV * NPT * 64 * 4);
  float*     x2     = (float*)take((size_t)B_EV * NPT * 128 * 4);
  unsigned*  pooled = (unsigned*)take((size_t)B_EV * 1024 * 4);
  _Float16*  c1W1h  = (_Float16*)take((size_t)32 * 64 * 2);    // K padded 6->32
  _Float16*  c1W2h  = (_Float16*)take((size_t)64 * 64 * 2);
  _Float16*  c1W3h  = (_Float16*)take((size_t)64 * 64 * 2);
  _Float16*  c2Wh   = (_Float16*)take((size_t)128 * 128 * 2);
  _Float16*  lWh    = (_Float16*)take((size_t)192 * 1024 * 2);

  // 0) init pooled keys (encodes -inf)
  dg_zero_u32<<<dim3((B_EV * 1024 + 255) / 256), dim3(256), 0, stream>>>(pooled, B_EV * 1024);

  // 1) weight f32 -> f16 conversions into WMMA fragment order
  dg_cvt_f16_sw<<<dim3((32 * 64 + 255) / 256),     dim3(256), 0, stream>>>(c1W1, c1W1h,   6,  32,   64);
  dg_cvt_f16_sw<<<dim3((64 * 64 + 255) / 256),     dim3(256), 0, stream>>>(c1W2, c1W2h,  64,  64,   64);
  dg_cvt_f16_sw<<<dim3((64 * 64 + 255) / 256),     dim3(256), 0, stream>>>(c1W3, c1W3h,  64,  64,   64);
  dg_cvt_f16_sw<<<dim3((128 * 128 + 255) / 256),   dim3(256), 0, stream>>>(c2W,  c2Wh,  128, 128,  128);
  dg_cvt_f16_sw<<<dim3((192 * 1024 + 255) / 256),  dim3(256), 0, stream>>>(lW,   lWh,   192, 192, 1024);

  // 2) kNN on pos
  dg_knn3<<<dim3(B_EV * 8), dim3(256), 0, stream>>>(pos, idx1);

  // 3) edge conv 1 -> x1 [B,N,64]
  dg_edgeconv1<<<dim3(B_EV * NPT / 4), dim3(128), 0, stream>>>(
      pos, idx1, c1W1h, c1b1, c1g1, c1t1, c1W2h, c1b2, c1g2, c1t2,
      c1W3h, c1b3, c1g3, c1t3, x1);

  // 4) kNN on x1 (WMMA distance matrix)
  dg_knn64w<<<dim3(B_EV * 16), dim3(256), 0, stream>>>(x1, idx2);

  // 5) edge conv 2 -> x2 [B,N,128]
  dg_edgeconv2<<<dim3(B_EV * NPT / 2), dim3(64), 0, stream>>>(
      x1, idx2, c2Wh, c2b, c2g, c2t, x2);

  // 6) lin1 + fused global max pool
  dg_lin1pool<<<dim3(8, NPT / 16, B_EV), dim3(256), 0, stream>>>(
      x1, x2, lWh, lb, lg, lt, pooled);

  // 7) head MLPs -> out [B,5]
  dg_head<<<dim3(B_EV), dim3(256), 0, stream>>>(
      pooled, m1W, m1b, m1g, m1t, m2W, m2b, m2g, m2t, hW, hb, out);
}
